// pLoss_51788715655587
// MI455X (gfx1250) — compile-verified, matrix operand order checked by reading, not run on hardware
//
#include <hip/hip_runtime.h>
#include <hip/hip_bf16.h>

typedef __attribute__((ext_vector_type(16))) _Float16 v16h;
typedef __attribute__((ext_vector_type(8)))  _Float16 v8h;
typedef __attribute__((ext_vector_type(8)))  float    v8f;
typedef __attribute__((ext_vector_type(4)))  float    v4f;

#define N_STATES 2048
#define N_LABELS 32
#define BATCH    32768
#define WAVES_PER_BLOCK 8
#define BATCH_PER_WAVE  16
#define BATCH_PER_BLOCK (WAVES_PER_BLOCK * BATCH_PER_WAVE)   // 128
#define LOG_CLAMP -100.0f

__global__ void pLoss_zero_kernel(float* out) { out[0] = 0.0f; }

__launch_bounds__(256, 1)
__global__ void pLoss_main_kernel(const float* __restrict__ f,
                                  const float* __restrict__ y,
                                  const float* __restrict__ mask,
                                  const float* __restrict__ S,
                                  float* __restrict__ out)
{
    // S staged in LDS as f16 in BOTH orientations:
    //   S_h[state][label]  -> A-fragments for potential = S @ f.T
    //   S_t[label][state]  -> B-fragments for pMargin_num = J.T @ S
    __shared__ __align__(16) _Float16 S_h[N_STATES * N_LABELS];   // 128 KB
    __shared__ __align__(16) _Float16 S_t[N_LABELS * N_STATES];   // 128 KB
    __shared__ float wsum[WAVES_PER_BLOCK];

    const int tid  = threadIdx.x;
    const int lane = tid & 31;
    const int wave = tid >> 5;
    const int l16  = lane & 15;
    const int hi   = lane >> 4;          // 0 for lanes 0-15, 1 for lanes 16-31

    // ---- Stage 0: cooperative f32 -> f16 conversion of S into LDS ----
    for (int idx = tid; idx < N_STATES * N_LABELS; idx += 256) {
        const int s = idx >> 5, l = idx & 31;
        const _Float16 v = (_Float16)S[idx];
        S_h[idx] = v;
        S_t[l * N_STATES + s] = v;
    }
    __syncthreads();

    const int b_base = blockIdx.x * BATCH_PER_BLOCK + wave * BATCH_PER_WAVE;

    // ---- B-fragment of f.T (K=32 labels x N=16 batch), held all kernel ----
    // B layout: lane n (0-15) = column n, K in halves {0..7,16..23};
    //           lane n+16     = column n, K in halves {8..15,24..31}.
    v16h bf;
    {
        const float* fp = f + (b_base + l16) * N_LABELS + hi * 8;
        const v4f f0 = *(const v4f*)(fp + 0);
        const v4f f1 = *(const v4f*)(fp + 4);
        const v4f f2 = *(const v4f*)(fp + 16);
        const v4f f3 = *(const v4f*)(fp + 20);
        #pragma unroll
        for (int i = 0; i < 4; ++i) {
            bf[i]      = (_Float16)f0[i];
            bf[4 + i]  = (_Float16)f1[i];
            bf[8 + i]  = (_Float16)f2[i];
            bf[12 + i] = (_Float16)f3[i];
        }
    }

    // A-fragment of one 16-state chunk of S (M=16 states x K=32 labels)
    auto load_sA = [&](int chunk) -> v16h {
        const _Float16* p = S_h + (chunk * 16 + l16) * N_LABELS + hi * 8;
        const v8h lo  = *(const v8h*)(p);        // labels base+0..7   (K 0..7 / 8..15)
        const v8h hi8 = *(const v8h*)(p + 16);   // labels base+16..23 (K 16..23 / 24..31)
        v16h a;
        #pragma unroll
        for (int i = 0; i < 8; ++i) { a[i] = lo[i]; a[8 + i] = hi8[i]; }
        return a;
    };

    // B-fragment of S for GEMM2: K=32 states (chunk pair) x N=16 labels (tile t)
    auto load_sB = [&](int t, int pair) -> v16h {
        const _Float16* p = S_t + (t * 16 + l16) * N_STATES + pair * 32 + hi * 8;
        const v8h lo  = *(const v8h*)(p);        // states base+0..7
        const v8h hi8 = *(const v8h*)(p + 16);   // states base+16..23
        v16h b;
        #pragma unroll
        for (int i = 0; i < 8; ++i) { b[i] = lo[i]; b[8 + i] = hi8[i]; }
        return b;
    };

    // ---- Pass A: running max over all 2048 states per batch column ----
    float m = -3.0e38f;
    for (int c = 0; c < N_STATES / 16; ++c) {
        const v16h a = load_sA(c);
        v8f C = {};
        C = __builtin_amdgcn_wmma_f32_16x16x32_f16(false, a, false, bf,
                                                   (short)0, C, false, false);
        #pragma unroll
        for (int j = 0; j < 8; ++j) m = fmaxf(m, C[j]);
    }
    m = fmaxf(m, __shfl_xor(m, 16, 32));   // combine row-halves: full column max

    // ---- Pass B: recompute potential, J = exp(.-m), z, and J.T @ S ----
    float zl = 0.0f;
    v8f acc0 = {};   // labels 0..15
    v8f acc1 = {};   // labels 16..31
    for (int p = 0; p < N_STATES / 32; ++p) {
        const v16h a0 = load_sA(2 * p);
        const v16h a1 = load_sA(2 * p + 1);
        v8f C0 = {}, C1 = {};
        C0 = __builtin_amdgcn_wmma_f32_16x16x32_f16(false, a0, false, bf,
                                                    (short)0, C0, false, false);
        C1 = __builtin_amdgcn_wmma_f32_16x16x32_f16(false, a1, false, bf,
                                                    (short)0, C1, false, false);
        // C-fragment -> A-fragment of J.T is lane-local (see analysis):
        //   halves 0..7  = exp(C0 rows) -> K {0..7 | 8..15}
        //   halves 8..15 = exp(C1 rows) -> K {16..23 | 24..31}
        v16h J;
        #pragma unroll
        for (int j = 0; j < 8; ++j) {
            const float e0 = __expf(C0[j] - m);
            const float e1 = __expf(C1[j] - m);
            zl += e0 + e1;
            J[j]     = (_Float16)e0;
            J[8 + j] = (_Float16)e1;
        }
        const v16h sb0 = load_sB(0, p);
        const v16h sb1 = load_sB(1, p);
        acc0 = __builtin_amdgcn_wmma_f32_16x16x32_f16(false, J, false, sb0,
                                                      (short)0, acc0, false, false);
        acc1 = __builtin_amdgcn_wmma_f32_16x16x32_f16(false, J, false, sb1,
                                                      (short)0, acc1, false, false);
    }
    zl += __shfl_xor(zl, 16, 32);          // full softmax denominator per column

    // ---- Epilogue: pMargin = acc/z, masked BCE, store ----
    float lsum = 0.0f;
    #pragma unroll
    for (int j = 0; j < 8; ++j) {
        const int   bl = j + hi * 8;                 // local batch row of this C slot
        const float rz = 1.0f / __shfl(zl, bl, 32);  // z[bl] broadcast from lane bl
        const int   gb = b_base + bl;

        const float p0 = acc0[j] * rz;
        const float p1 = acc1[j] * rz;

        const float y0 = y[gb * 32 + l16];
        const float y1 = y[gb * 32 + 16 + l16];
        const float m0 = mask[gb * 32 + l16];
        const float m1 = mask[gb * 32 + 16 + l16];

        const float lp0  = fmaxf(__logf(p0),        LOG_CLAMP);
        const float l1p0 = fmaxf(__logf(1.0f - p0), LOG_CLAMP);
        const float lp1  = fmaxf(__logf(p1),        LOG_CLAMP);
        const float l1p1 = fmaxf(__logf(1.0f - p1), LOG_CLAMP);

        lsum += m0 * -(y0 * lp0 + (1.0f - y0) * l1p0);
        lsum += m1 * -(y1 * lp1 + (1.0f - y1) * l1p1);

        out[1 + gb * 32 + l16]      = p0;
        out[1 + gb * 32 + 16 + l16] = p1;
    }

    // ---- Loss reduction: wave butterfly -> block -> one atomic per block ----
    #pragma unroll
    for (int off = 16; off > 0; off >>= 1) lsum += __shfl_xor(lsum, off, 32);
    if (lane == 0) wsum[wave] = lsum;
    __syncthreads();
    if (tid == 0) {
        float t = 0.0f;
        #pragma unroll
        for (int w = 0; w < WAVES_PER_BLOCK; ++w) t += wsum[w];
        atomicAdd(out, t * (1.0f / (float)BATCH));
    }
}

extern "C" void kernel_launch(void* const* d_in, const int* in_sizes, int n_in,
                              void* d_out, int out_size, void* d_ws, size_t ws_size,
                              hipStream_t stream) {
    const float* f    = (const float*)d_in[0];
    const float* y    = (const float*)d_in[1];
    const float* mask = (const float*)d_in[2];
    const float* S    = (const float*)d_in[3];
    float* out = (float*)d_out;

    pLoss_zero_kernel<<<1, 1, 0, stream>>>(out);
    pLoss_main_kernel<<<BATCH / BATCH_PER_BLOCK, 256, 0, stream>>>(f, y, mask, S, out);
}
